// ChunkedDynamicBlock_10788957847689
// MI455X (gfx1250) — matrix operand order repaired and checked by reading, not compiled, not run on hardware
//
#include <hip/hip_runtime.h>

// ---------------- problem constants ----------------
#define BATCH       16384
#define IN_F        4096
#define NUM_CHUNKS  8
#define CHUNK       512     // CHUNK_IN == CHUNK_OUT == 512
#define TOPK        2
#define OUT_BASE    ((size_t)BATCH * TOPK * CHUNK)   // 16384*1024

// WMMA vector types (CDNA5, wave32)
typedef __attribute__((ext_vector_type(16))) __bf16 bf16x16;
typedef __attribute__((ext_vector_type(8)))  float  f32x8;

// ---------------- ws layout ----------------
// [0, 16KB)      : 4096 float partial sums  (8 chunks x 512 rowblocks)
// [16KB, +8B)    : 2 int topk indices
// [32KB, +2MB)   : bf16 B-fragments, layout:
//   frag = t*512 + kstep*32 + ntile          (1024 bf16 each)
//   within frag: [hi: 32 lanes x 16 bf16][lo: 32 lanes x 16 bf16]
//   lane L holds B[kbase + j][n], j=0..15, kbase=(L>>4)*16, n=L&15  (contiguous k order, same as A)
#define WS_PART_OFF 0
#define WS_TOPK_OFF 16384
#define WS_B_OFF    32768
#define FRAG_ELEMS  1024

// =============== kernel 1: per-(chunk,rowblock) partial |x| sums ===============
__global__ __launch_bounds__(256)
void act_partial_kernel(const float* __restrict__ x, float* __restrict__ ws_part) {
    const int rb  = blockIdx.x;          // 0..511  (32 rows each)
    const int c   = blockIdx.y;          // 0..7    chunk
    const int tid = threadIdx.x;
    const float* base = x + ((size_t)rb * 32) * IN_F + (size_t)c * CHUNK;
    float s = 0.f;
#pragma unroll
    for (int it = 0; it < 16; ++it) {
        int e  = it * 256 + tid;         // float4 index, 0..4095 (32 rows * 128)
        int r  = e >> 7;
        int c4 = e & 127;
        float4 v = ((const float4*)(base + (size_t)r * IN_F))[c4];
        s += fabsf(v.x) + fabsf(v.y) + fabsf(v.z) + fabsf(v.w);
    }
    __shared__ float red[256];
    red[tid] = s; __syncthreads();
#pragma unroll
    for (int off = 128; off > 0; off >>= 1) {
        if (tid < off) red[tid] += red[tid + off];
        __syncthreads();
    }
    if (tid == 0) ws_part[c * 512 + rb] = red[0];
}

// =============== kernel 2: finalize activities + top-2 (deterministic) ===============
__global__ __launch_bounds__(256)
void finalize_kernel(const float* __restrict__ ws_part,
                     float* __restrict__ out_act,      // d_out + OUT_BASE (10 floats)
                     int* __restrict__ ws_topk) {
    const int tid = threadIdx.x;
    const int c = tid >> 5;          // chunk 0..7
    const int j = tid & 31;
    float s = 0.f;
    for (int q = j; q < 512; q += 32) s += ws_part[c * 512 + q];
    __shared__ float red[256];
    red[tid] = s; __syncthreads();
#pragma unroll
    for (int off = 16; off > 0; off >>= 1) {
        if ((tid & 31) < off) red[tid] += red[tid + off];
        __syncthreads();
    }
    __shared__ float acts[8];
    if ((tid & 31) == 0) acts[c] = red[c * 32] * (1.0f / ((float)BATCH * (float)CHUNK));
    __syncthreads();
    if (tid == 0) {
        for (int q = 0; q < 8; ++q) out_act[q] = acts[q];
        int i0 = 0;
        for (int q = 1; q < 8; ++q) if (acts[q] > acts[i0]) i0 = q;     // ties -> lowest idx
        int i1 = -1;
        for (int q = 0; q < 8; ++q) {
            if (q == i0) continue;
            if (i1 < 0 || acts[q] > acts[i1]) i1 = q;
        }
        out_act[8] = (float)i0;
        out_act[9] = (float)i1;
        ws_topk[0] = i0;
        ws_topk[1] = i1;
    }
}

// =============== kernel 3: W -> bf16 hi/lo pre-swizzled B-fragments ===============
__global__ __launch_bounds__(256)
void prep_w_kernel(const float* __restrict__ W, const int* __restrict__ topk,
                   __bf16* __restrict__ wsB) {
    unsigned u = blockIdx.x * 256u + threadIdx.x;   // < 524288
    const int i     = u & 15;
    const int lane  = (u >> 4) & 31;
    const int ntile = (u >> 9) & 31;
    const int ks    = (u >> 14) & 15;
    const int t     = (u >> 18) & 1;
    const int k     = topk[t];
    const int K = ks * 32 + (lane >> 4) * 16 + i;   // B-fragment K, contiguous k order (same as A)
    const int N = ntile * 16 + (lane & 15);
    const float wv = W[((size_t)k * CHUNK + K) * CHUNK + N];
    const __bf16 h = (__bf16)wv;
    const __bf16 l = (__bf16)(wv - (float)h);
    const size_t base = ((size_t)t * 512 + (size_t)ks * 32 + ntile) * FRAG_ELEMS
                        + (size_t)lane * 16 + i;
    wsB[base]       = h;          // hi half
    wsB[base + 512] = l;          // lo half
}

// =============== kernel 4: split-bf16 WMMA GEMM ===============
// block = 512 threads = 16 waves; block tile = 64 rows x 512 cols of one chunk.
// wave w: mtile = w>>2 (16 rows), ngroup = w&3 (8 ntiles of 16 cols).
#define GEMM_ROWS 64
#define SLABW     516          // 512 + 4 pad floats: breaks 2KB-stride bank conflicts

__global__ __launch_bounds__(512)
void gemm_kernel(const float* __restrict__ x, const float* __restrict__ bias,
                 const int* __restrict__ topk, const __bf16* __restrict__ wsB,
                 float* __restrict__ out) {
    __shared__ float slab[GEMM_ROWS][SLABW];        // 132,096 B (<=320KB/WGP)

    const int t     = blockIdx.y;                   // selected-chunk slot 0/1
    const int row0  = blockIdx.x * GEMM_ROWS;
    const int k_sel = topk[t];
    const int tid   = threadIdx.x;

    // ---- cooperative, coalesced load of the x slab (64 rows x 512 f32) ----
    const float* xbase = x + (size_t)row0 * IN_F + (size_t)k_sel * CHUNK;
#pragma unroll
    for (int it = 0; it < 16; ++it) {
        int e  = it * 512 + tid;       // float4 index, 0..8191
        int r  = e >> 7;
        int c4 = e & 127;
        float4 v = ((const float4*)(xbase + (size_t)r * IN_F))[c4];
        ((float4*)&slab[r][0])[c4] = v;
    }
    __syncthreads();

    const int w      = tid >> 5;
    const int lane   = tid & 31;
    const int mtile  = w >> 2;         // 0..3
    const int ngroup = w & 3;          // 0..3 -> ntiles ngroup*8 .. +7
    const int m      = mtile * 16 + (lane & 15);
    const int g      = lane >> 4;

    f32x8 acc[8];
    const f32x8 zero = {};
#pragma unroll
    for (int nt = 0; nt < 8; ++nt) acc[nt] = zero;

    const __bf16* wsBt = wsB + (size_t)t * 512 * FRAG_ELEMS;

#pragma unroll 1
    for (int ks = 0; ks < 16; ++ks) {
        // ---- build A hi/lo fragments from LDS (contiguous k: lane half g holds k = 16g..16g+15) ----
        const int c0 = ks * 32 + g * 16;
        const float* arow = &slab[m][0];
        float4 f0 = *(const float4*)(arow + c0);
        float4 f1 = *(const float4*)(arow + c0 + 4);
        float4 f2 = *(const float4*)(arow + c0 + 8);
        float4 f3 = *(const float4*)(arow + c0 + 12);
        float af[16] = { f0.x, f0.y, f0.z, f0.w,  f1.x, f1.y, f1.z, f1.w,
                         f2.x, f2.y, f2.z, f2.w,  f3.x, f3.y, f3.z, f3.w };
        bf16x16 ahi, alo;
#pragma unroll
        for (int i = 0; i < 16; ++i) {
            __bf16 h = (__bf16)af[i];
            ahi[i] = h;
            alo[i] = (__bf16)(af[i] - (float)h);
        }

        const __bf16* fb = wsBt + ((size_t)ks * 32 + (size_t)ngroup * 8) * FRAG_ELEMS;
        if (ks < 15) {  // exercise global_prefetch on next K-step's B fragments
            __builtin_prefetch(wsBt + ((size_t)(ks + 1) * 32 + (size_t)ngroup * 8) * FRAG_ELEMS, 0, 0);
        }

#pragma unroll
        for (int nt = 0; nt < 8; ++nt) {
            const __bf16* ph = fb + (size_t)nt * FRAG_ELEMS + (size_t)lane * 16;
            const __bf16* pl = ph + 512;
            union { uint4 u[2]; bf16x16 v; } Bh, Bl;
            Bh.u[0] = *(const uint4*)(ph);
            Bh.u[1] = *(const uint4*)(ph + 8);
            Bl.u[0] = *(const uint4*)(pl);
            Bl.u[1] = *(const uint4*)(pl + 8);
            // x*w ~= xhi*whi + xhi*wlo + xlo*whi  (near-f32 accuracy, D->C chain: no hazard)
            acc[nt] = __builtin_amdgcn_wmma_f32_16x16x32_bf16(
                          false, ahi, false, Bh.v, (short)0, acc[nt], false, false);
            acc[nt] = __builtin_amdgcn_wmma_f32_16x16x32_bf16(
                          false, ahi, false, Bl.v, (short)0, acc[nt], false, false);
            acc[nt] = __builtin_amdgcn_wmma_f32_16x16x32_bf16(
                          false, alo, false, Bh.v, (short)0, acc[nt], false, false);
        }
    }

    // ---- epilogue: bias add + store (C/D layout: VGPR r -> M = 8*g + r, lane -> N) ----
    const int n_      = lane & 15;
    const int rowbase = row0 + mtile * 16 + g * 8;
#pragma unroll
    for (int nt = 0; nt < 8; ++nt) {
        const int ntile = ngroup * 8 + nt;
        const int ocol  = ntile * 16 + n_;
        const float bv  = bias[(size_t)k_sel * CHUNK + ocol];
        float* op = out + (size_t)rowbase * (TOPK * CHUNK) + (size_t)t * CHUNK + ocol;
#pragma unroll
        for (int r = 0; r < 8; ++r) {
            op[(size_t)r * (TOPK * CHUNK)] = acc[nt][r] + bv;
        }
    }
}

// =============== host launcher ===============
extern "C" void kernel_launch(void* const* d_in, const int* in_sizes, int n_in,
                              void* d_out, int out_size, void* d_ws, size_t ws_size,
                              hipStream_t stream) {
    const float* x    = (const float*)d_in[0];   // (16384, 4096) f32
    const float* W    = (const float*)d_in[1];   // (8, 512, 512) f32
    const float* bias = (const float*)d_in[2];   // (8, 512) f32
    float* out = (float*)d_out;                  // 16384*1024 + 8 + 2 floats

    float*  ws_part = (float*)((char*)d_ws + WS_PART_OFF);
    int*    ws_topk = (int*)  ((char*)d_ws + WS_TOPK_OFF);
    __bf16* wsB     = (__bf16*)((char*)d_ws + WS_B_OFF);   // needs ~2.1 MB of ws

    act_partial_kernel<<<dim3(512, 8), 256, 0, stream>>>(x, ws_part);
    finalize_kernel<<<1, 256, 0, stream>>>(ws_part, out + OUT_BASE, ws_topk);
    prep_w_kernel<<<2048, 256, 0, stream>>>(W, ws_topk, wsB);
    gemm_kernel<<<dim3(BATCH / GEMM_ROWS, TOPK), 512, 0, stream>>>(x, bias, ws_topk, wsB, out);
}
